// BiggerBirdBartForSequenceClassification_19224273617381
// MI455X (gfx1250) — matrix operand order repaired and checked by reading, not compiled, not run on hardware
//
#include <hip/hip_runtime.h>
#include <math.h>

// Problem constants (match reference)
#define T_SEQ   2048
#define D_MODEL 768
#define N_HEADS 12
#define HEAD_D  64
#define BATCH   2
#define BHTOT   24      // B * H
#define FRAGW   16
#define KSEL    6
#define NPROTO  32
#define TOPU    12
#define NGLOB   4

typedef _Float16 v16h __attribute__((ext_vector_type(16)));
typedef _Float16 h8   __attribute__((ext_vector_type(8)));
typedef float    v8f  __attribute__((ext_vector_type(8)));

static __device__ __forceinline__ v16h cat8(h8 lo, h8 hi) {
  v16h r;
#pragma unroll
  for (int i = 0; i < 8; ++i) { r[i] = lo[i]; r[i + 8] = hi[i]; }
  return r;
}

// ======================================================================
// One-shot f32 -> f16 conversion (done once per tensor instead of
// per-tile inside the GEMM hot loop).  n must be a multiple of 8.
// ======================================================================
__global__ __launch_bounds__(256)
void cvt_f32_to_f16_kernel(const float* __restrict__ src, _Float16* __restrict__ dst, int n) {
  const int i = (blockIdx.x * 256 + threadIdx.x) * 8;
  if (i >= n) return;
  float4 v0 = *(const float4*)(src + i);
  float4 v1 = *(const float4*)(src + i + 4);
  h8 h;
  h[0] = (_Float16)v0.x; h[1] = (_Float16)v0.y; h[2] = (_Float16)v0.z; h[3] = (_Float16)v0.w;
  h[4] = (_Float16)v1.x; h[5] = (_Float16)v1.y; h[6] = (_Float16)v1.z; h[7] = (_Float16)v1.w;
  *(h8*)(dst + i) = h;
}

// ======================================================================
// WMMA GEMM:  C[M,N] = (A[M,K] * W[N,K]^T + bias[N]) * scale
// A, W row-major f16 (pre-converted); fp32 accumulation via
// v_wmma_f32_16x16x32_f16.  Block tile 128x128, BK=32, 256 threads =
// 8 waves; each wave owns 32x64 = 2x4 fragments (8 WMMA / K-step).
// ======================================================================
#define TM 128
#define TN 128
#define TK 32
#define LDS_STRIDE 48   // halfs per LDS row (32 data + 16 pad, keeps 16B align)

__global__ __launch_bounds__(256)
void gemm_nt_bias_wmma(const _Float16* __restrict__ A, const _Float16* __restrict__ W,
                       const float* __restrict__ bias, float* __restrict__ C,
                       int M, int N, int K, float scale) {
  __shared__ __align__(16) _Float16 As[TM * LDS_STRIDE];
  __shared__ __align__(16) _Float16 Bs[TN * LDS_STRIDE];

  const int tid   = threadIdx.x;
  const int lane  = tid & 31;
  const int wave  = tid >> 5;
  const int wm    = wave & 3;   // 4 waves along M (32 rows each)
  const int wn    = wave >> 2;  // 2 waves along N (64 cols each)
  const int lrow  = lane & 15;
  const int lhalf = lane >> 4;
  const int m0    = blockIdx.x * TM;
  const int n0    = blockIdx.y * TN;

  v8f acc[2][4] = {};

  const int ksteps = K / TK;
  for (int kt = 0; kt < ksteps; ++kt) {
    const int k0 = kt * TK;
    // ---- stage A and B tiles: 128 rows x 32 halfs each; pure copy ----
    {
      const int r  = tid >> 1;
      const int c0 = (tid & 1) * 16;
      const _Float16* srcA = &A[(size_t)(m0 + r) * K + k0 + c0];
      h8 a0 = *(const h8*)(srcA);
      h8 a1 = *(const h8*)(srcA + 8);
      *(h8*)&As[r * LDS_STRIDE + c0]     = a0;
      *(h8*)&As[r * LDS_STRIDE + c0 + 8] = a1;
      const _Float16* srcB = &W[(size_t)(n0 + r) * K + k0 + c0];
      h8 b0 = *(const h8*)(srcB);
      h8 b1 = *(const h8*)(srcB + 8);
      *(h8*)&Bs[r * LDS_STRIDE + c0]     = b0;
      *(h8*)&Bs[r * LDS_STRIDE + c0 + 8] = b1;
      if (kt + 1 < ksteps) {   // CDNA5 prefetch of next K tile (global_prefetch_b8)
        __builtin_prefetch(srcA + TK, 0, 1);
        __builtin_prefetch(srcB + TK, 0, 1);
      }
    }
    __syncthreads();

    // ---- fragment loads per ISA 7.12.2 layouts ----
    v16h afrag[2], bfrag[4];
#pragma unroll
    for (int fm = 0; fm < 2; ++fm) {
      const int r  = wm * 32 + fm * 16 + lrow;
      const int cb = lhalf * 8;   // lanes<16: K{0..7,16..23}; lanes>=16: K{8..15,24..31}
      h8 lo = *(const h8*)&As[r * LDS_STRIDE + cb];
      h8 hi = *(const h8*)&As[r * LDS_STRIDE + cb + 16];
      afrag[fm] = cat8(lo, hi);
    }
#pragma unroll
    for (int fn = 0; fn < 4; ++fn) {
      const int n  = wn * 64 + fn * 16 + lrow;
      const int cb = lhalf * 8;   // lanes<16: K{0..7,16..23}; lanes>=16: K{8..15,24..31}
      h8 lo = *(const h8*)&Bs[n * LDS_STRIDE + cb];
      h8 hi = *(const h8*)&Bs[n * LDS_STRIDE + cb + 16];
      bfrag[fn] = cat8(lo, hi);
    }
#pragma unroll
    for (int fm = 0; fm < 2; ++fm)
#pragma unroll
      for (int fn = 0; fn < 4; ++fn)
        acc[fm][fn] = __builtin_amdgcn_wmma_f32_16x16x32_f16(
            false, afrag[fm], false, bfrag[fn], (short)0, acc[fm][fn], false, false);
    __syncthreads();
  }

  // ---- epilogue: C VGPR i -> M = i + 8*lhalf, N = lrow ----
#pragma unroll
  for (int fm = 0; fm < 2; ++fm) {
#pragma unroll
    for (int fn = 0; fn < 4; ++fn) {
      const int n  = n0 + wn * 64 + fn * 16 + lrow;
      const float bv = bias[n];
#pragma unroll
      for (int i = 0; i < 8; ++i) {
        const int m = m0 + wm * 32 + fm * 16 + lhalf * 8 + i;
        C[(size_t)m * N + n] = (acc[fm][fn][i] + bv) * scale;
      }
    }
  }
}

// ======================================================================
// Local windowed MMR selection: one block = (bh, 64 consecutive t)
// Stages <=80 key rows + norms in LDS, each thread runs 6-round select.
// ======================================================================
#define KW_STRIDE 68

__global__ __launch_bounds__(64)
void local_select_kernel(const float* __restrict__ qbuf, const float* __restrict__ kbuf,
                         int* __restrict__ sel) {
  const int bh = blockIdx.x;
  const int t0 = blockIdx.y * 64;
  const int b  = bh / N_HEADS, h = bh % N_HEADS;

  __shared__ float kw[80 * KW_STRIDE];
  __shared__ float nrm[80];

  int minStart = t0 - 8; if (minStart < 0) minStart = 0; if (minStart > T_SEQ - FRAGW) minStart = T_SEQ - FRAGW;
  int base = minStart; if (base > T_SEQ - 80) base = T_SEQ - 80;

  for (int i = threadIdx.x; i < 80 * 16; i += 64) {
    const int r = i >> 4, c4 = (i & 15) << 2;
    float4 v = *(const float4*)&kbuf[((size_t)(b * T_SEQ + base + r)) * D_MODEL + h * HEAD_D + c4];
    kw[r * KW_STRIDE + c4 + 0] = v.x; kw[r * KW_STRIDE + c4 + 1] = v.y;
    kw[r * KW_STRIDE + c4 + 2] = v.z; kw[r * KW_STRIDE + c4 + 3] = v.w;
  }
  __syncthreads();
  for (int r = threadIdx.x; r < 80; r += 64) {
    float s = 0.f;
    for (int dd = 0; dd < HEAD_D; ++dd) { float a = kw[r * KW_STRIDE + dd]; s += a * a; }
    nrm[r] = fmaxf(sqrtf(s), 1e-6f);
  }
  __syncthreads();

  const int t = t0 + (int)threadIdx.x;
  int start = t - 8; if (start < 0) start = 0; if (start > T_SEQ - FRAGW) start = T_SEQ - FRAGW;
  const int off = start - base;

  float qv[HEAD_D];
  {
    const float* qr = &qbuf[((size_t)(b * T_SEQ + t)) * D_MODEL + h * HEAD_D];
#pragma unroll
    for (int dd = 0; dd < HEAD_D; ++dd) qv[dd] = qr[dd];
  }

  float sc[FRAGW];
  for (int f = 0; f < FRAGW; ++f) {
    const float* kr = &kw[(off + f) * KW_STRIDE];
    float dot = 0.f;
    for (int dd = 0; dd < HEAD_D; ++dd) dot += qv[dd] * kr[dd];
    const float dist = fabsf((float)(start + f - t));
    sc[f] = dot + 0.2f * __expf(-dist / 8.0f);
  }

  for (int r = 0; r < KSEL; ++r) {
    int j = 0; float best = sc[0];
    for (int f = 1; f < FRAGW; ++f) if (sc[f] > best) { best = sc[f]; j = f; }
    sel[((size_t)bh * T_SEQ + t) * KSEL + r] = start + j;
    if (r < KSEL - 1) {
      const float* kj = &kw[(off + j) * KW_STRIDE];
      const float nj = nrm[off + j];
      for (int f = 0; f < FRAGW; ++f) {
        const float* kf = &kw[(off + f) * KW_STRIDE];
        float dt = 0.f;
        for (int dd = 0; dd < HEAD_D; ++dd) dt += kf[dd] * kj[dd];
        float cs = dt / (nrm[off + f] * nj);
        if (cs < 0.f) cs = 0.f;
        sc[f] -= 0.2f * cs;
      }
    }
    sc[j] = -1e9f;
  }
}

// ======================================================================
// Qp prototypes: h' groups pair flat heads (2h', 2h'+1) per reference's
// reshape(H,B,T,d) quirk. One thread per (h', p).
// ======================================================================
__global__ void qp_kernel(const float* __restrict__ qbuf, float* __restrict__ Qp) {
  const int tid = threadIdx.x;
  if (tid >= N_HEADS * NPROTO) return;
  const int hp = tid / NPROTO, p = tid % NPROTO;
  const int idxp = __float2int_rn((float)p * 2047.0f / 31.0f);
  const int i0 = 2 * hp, i1 = 2 * hp + 1;
  const int b0 = i0 / N_HEADS, h0 = i0 % N_HEADS;
  const int b1 = i1 / N_HEADS, h1 = i1 % N_HEADS;
  const float* r0 = &qbuf[((size_t)(b0 * T_SEQ + idxp)) * D_MODEL + h0 * HEAD_D];
  const float* r1 = &qbuf[((size_t)(b1 * T_SEQ + idxp)) * D_MODEL + h1 * HEAD_D];
  float v[HEAD_D]; float s = 0.f;
#pragma unroll
  for (int dd = 0; dd < HEAD_D; ++dd) { v[dd] = 0.5f * (r0[dd] + r1[dd]); s += v[dd] * v[dd]; }
  const float inv = 1.0f / fmaxf(sqrtf(s), 1e-6f);
#pragma unroll
  for (int dd = 0; dd < HEAD_D; ++dd) Qp[(hp * NPROTO + p) * HEAD_D + dd] = v[dd] * inv;
}

// ======================================================================
// S = relu(Kbar . Qp) and utility u.  One thread per (h', t).
// ======================================================================
__global__ __launch_bounds__(256)
void proto_score_kernel(const float* __restrict__ kbuf, const float* __restrict__ Qp,
                        float* __restrict__ S, float* __restrict__ u) {
  const int hp = blockIdx.x;
  const int t  = blockIdx.y * 256 + threadIdx.x;
  __shared__ float qp[NPROTO * HEAD_D];
  for (int i = threadIdx.x; i < NPROTO * HEAD_D; i += 256) qp[i] = Qp[hp * NPROTO * HEAD_D + i];
  __syncthreads();

  const int i0 = 2 * hp, i1 = 2 * hp + 1;
  const int b0 = i0 / N_HEADS, h0 = i0 % N_HEADS;
  const int b1 = i1 / N_HEADS, h1 = i1 % N_HEADS;
  const float* r0 = &kbuf[((size_t)(b0 * T_SEQ + t)) * D_MODEL + h0 * HEAD_D];
  const float* r1 = &kbuf[((size_t)(b1 * T_SEQ + t)) * D_MODEL + h1 * HEAD_D];
  float kb[HEAD_D]; float nn = 0.f;
#pragma unroll
  for (int dd = 0; dd < HEAD_D; ++dd) { kb[dd] = 0.5f * (r0[dd] + r1[dd]); nn += kb[dd] * kb[dd]; }
  const float inv = 1.0f / fmaxf(sqrtf(nn), 1e-6f);
#pragma unroll
  for (int dd = 0; dd < HEAD_D; ++dd) kb[dd] *= inv;

  float s[NPROTO];
  float sum = 0.f, sumsq = 0.f, mx = -3.4e38f;
  for (int p = 0; p < NPROTO; ++p) {
    float dot = 0.f;
    for (int dd = 0; dd < HEAD_D; ++dd) dot += kb[dd] * qp[p * HEAD_D + dd];
    if (dot < 0.f) dot = 0.f;
    s[p] = dot; sum += dot; sumsq += dot * dot;
    if (dot > mx) mx = dot;
    S[((size_t)hp * T_SEQ + t) * NPROTO + p] = dot;
  }
  const float mean = sum / (float)NPROTO;
  float var = (sumsq - (float)NPROTO * mean * mean) / (float)(NPROTO - 1);
  if (var < 0.f) var = 0.f;
  // mean of top-6
  float tmp[NPROTO];
  for (int p = 0; p < NPROTO; ++p) tmp[p] = s[p];
  float topsum = 0.f;
  for (int r = 0; r < 6; ++r) {
    int j = 0; float best = tmp[0];
    for (int p = 1; p < NPROTO; ++p) if (tmp[p] > best) { best = tmp[p]; j = p; }
    topsum += best; tmp[j] = -1e9f;
  }
  u[hp * T_SEQ + t] = 1.0f * mean + 0.6f * mx + 0.4f * (topsum / 6.0f) + 0.2f * sqrtf(var);
}

// ======================================================================
// Per-head top-12 + greedy facility selection -> 4 global key indices.
// ======================================================================
__global__ __launch_bounds__(256)
void greedy_kernel(const float* __restrict__ u, const float* __restrict__ S,
                   int* __restrict__ glob) {
  const int hp = blockIdx.x;
  __shared__ float su[T_SEQ];
  __shared__ float bval[256];
  __shared__ int   bidx[256];
  __shared__ int   top_idx[TOPU];
  __shared__ float Ssub[TOPU * NPROTO];

  for (int i = threadIdx.x; i < T_SEQ; i += 256) su[i] = u[hp * T_SEQ + i];
  __syncthreads();

  for (int r = 0; r < TOPU; ++r) {
    float best = -3.4e38f; int besti = 0x7fffffff;
    for (int i = threadIdx.x; i < T_SEQ; i += 256) {
      float v = su[i];
      if (v > best || (v == best && i < besti)) { best = v; besti = i; }
    }
    bval[threadIdx.x] = best; bidx[threadIdx.x] = besti;
    __syncthreads();
    for (int s = 128; s > 0; s >>= 1) {
      if ((int)threadIdx.x < s) {
        float v2 = bval[threadIdx.x + s]; int i2 = bidx[threadIdx.x + s];
        if (v2 > bval[threadIdx.x] || (v2 == bval[threadIdx.x] && i2 < bidx[threadIdx.x])) {
          bval[threadIdx.x] = v2; bidx[threadIdx.x] = i2;
        }
      }
      __syncthreads();
    }
    if (threadIdx.x == 0) { top_idx[r] = bidx[0]; su[bidx[0]] = -1e9f; }
    __syncthreads();
  }

  for (int i = threadIdx.x; i < TOPU * NPROTO; i += 256) {
    const int row = i / NPROTO, p = i % NPROTO;
    Ssub[i] = S[((size_t)hp * T_SEQ + top_idx[row]) * NPROTO + p];
  }
  __syncthreads();

  if (threadIdx.x == 0) {
    float m[NPROTO];
    bool blocked[TOPU];
#pragma unroll
    for (int p = 0; p < NPROTO; ++p) m[p] = 0.f;
    for (int i = 0; i < TOPU; ++i) blocked[i] = false;
    for (int r = 0; r < NGLOB; ++r) {
      int j = 0; float best = -3.4e38f;
      for (int i = 0; i < TOPU; ++i) {
        float g;
        if (blocked[i]) g = -1e9f;
        else {
          g = 0.f;
          for (int p = 0; p < NPROTO; ++p) {
            float d = Ssub[i * NPROTO + p] - m[p];
            if (d > 0.f) g += d;
          }
        }
        if (g > best) { best = g; j = i; }
      }
      glob[hp * NGLOB + r] = top_idx[j];
      blocked[j] = true;
      for (int p = 0; p < NPROTO; ++p) {
        float v = Ssub[j * NPROTO + p];
        if (v > m[p]) m[p] = v;
      }
    }
  }
}

// ======================================================================
// Sparse attention over 10 candidates (6 local + 4 global) per (bh,t).
// Context written in (B,T,D) layout directly as f16 so the Wo WMMA
// projection consumes it with no extra conversion pass.
// ======================================================================
__global__ __launch_bounds__(64)
void attend_kernel(const float* __restrict__ qbuf, const float* __restrict__ kbuf,
                   const float* __restrict__ vbuf, const int* __restrict__ sel,
                   const int* __restrict__ glob, _Float16* __restrict__ att) {
  const int bh = blockIdx.x;
  const int t  = blockIdx.y * 64 + threadIdx.x;
  const int b  = bh / N_HEADS, h = bh % N_HEADS;

  int cand[KSEL + NGLOB];
#pragma unroll
  for (int c = 0; c < KSEL; ++c) cand[c] = sel[((size_t)bh * T_SEQ + t) * KSEL + c];
#pragma unroll
  for (int g = 0; g < NGLOB; ++g) cand[KSEL + g] = glob[h * NGLOB + g];

  float qv[HEAD_D];
  {
    const float* qr = &qbuf[((size_t)(b * T_SEQ + t)) * D_MODEL + h * HEAD_D];
#pragma unroll
    for (int dd = 0; dd < HEAD_D; ++dd) qv[dd] = qr[dd];
  }

  float lg[KSEL + NGLOB];
  float mx = -3.4e38f;
#pragma unroll
  for (int c = 0; c < KSEL + NGLOB; ++c) {
    const float* kr = &kbuf[((size_t)(b * T_SEQ + cand[c])) * D_MODEL + h * HEAD_D];
    float dot = 0.f;
    for (int dd = 0; dd < HEAD_D; ++dd) dot += qv[dd] * kr[dd];
    lg[c] = dot;
    if (dot > mx) mx = dot;
  }
  float o[HEAD_D];
#pragma unroll
  for (int dd = 0; dd < HEAD_D; ++dd) o[dd] = 0.f;
  float ssum = 0.f;
#pragma unroll
  for (int c = 0; c < KSEL + NGLOB; ++c) {
    const float w = __expf(lg[c] - mx);
    ssum += w;
    const float* vr = &vbuf[((size_t)(b * T_SEQ + cand[c])) * D_MODEL + h * HEAD_D];
    for (int dd = 0; dd < HEAD_D; ++dd) o[dd] += w * vr[dd];
  }
  const float inv = 1.0f / ssum;
  _Float16* ar = &att[((size_t)(b * T_SEQ + t)) * D_MODEL + h * HEAD_D];
#pragma unroll
  for (int dd = 0; dd < HEAD_D; ++dd) ar[dd] = (_Float16)(o[dd] * inv);
}

// ======================================================================
// Launch
// ======================================================================
extern "C" void kernel_launch(void* const* d_in, const int* in_sizes, int n_in,
                              void* d_out, int out_size, void* d_ws, size_t ws_size,
                              hipStream_t stream) {
  (void)in_sizes; (void)n_in; (void)out_size; (void)ws_size;
  const float* x  = (const float*)d_in[0];
  const float* Wq = (const float*)d_in[1];
  const float* bq = (const float*)d_in[2];
  const float* Wk = (const float*)d_in[3];
  const float* bk = (const float*)d_in[4];
  const float* Wv = (const float*)d_in[5];
  const float* bv = (const float*)d_in[6];
  const float* Wo = (const float*)d_in[7];
  const float* bo = (const float*)d_in[8];
  float* out = (float*)d_out;

  const int M = BATCH * T_SEQ;       // 4096
  const int N = D_MODEL, K = D_MODEL;
  const int NX = M * D_MODEL;        // x / q / k / v / att element count
  const int NW = D_MODEL * D_MODEL;  // weight element count

  char* ws = (char*)d_ws;
  const size_t SZF = (size_t)NX * sizeof(float);      // 12.58 MB
  const size_t SZH = (size_t)NX * sizeof(_Float16);   //  6.29 MB
  const size_t SZWH = (size_t)NW * sizeof(_Float16);  //  1.18 MB
  float*     qb  = (float*)(ws);
  float*     kb  = (float*)(ws + SZF);
  float*     vb  = (float*)(ws + 2 * SZF);
  _Float16*  xh  = (_Float16*)(ws + 3 * SZF);
  _Float16*  Wqh = (_Float16*)(ws + 3 * SZF + SZH);
  _Float16*  Wkh = (_Float16*)(ws + 3 * SZF + SZH + SZWH);
  _Float16*  Wvh = (_Float16*)(ws + 3 * SZF + SZH + 2 * SZWH);
  _Float16*  Woh = (_Float16*)(ws + 3 * SZF + SZH + 3 * SZWH);
  _Float16*  ah  = (_Float16*)(ws + 3 * SZF + SZH + 4 * SZWH);
  char*      p2  = ws + 3 * SZF + 2 * SZH + 4 * SZWH;
  int*       sel = (int*)(p2);
  float*     Qp  = (float*)(p2 + (size_t)BHTOT * T_SEQ * KSEL * sizeof(int));
  float*     S   = Qp + N_HEADS * NPROTO * HEAD_D;
  float*     u   = S + (size_t)N_HEADS * T_SEQ * NPROTO;
  int*      glob = (int*)(u + N_HEADS * T_SEQ);

  const float qscale = 0.125f;       // 64^-0.5

  // One-shot f16 conversions of GEMM operands.
  cvt_f32_to_f16_kernel<<<NX / (256 * 8), 256, 0, stream>>>(x,  xh,  NX);
  cvt_f32_to_f16_kernel<<<NW / (256 * 8), 256, 0, stream>>>(Wq, Wqh, NW);
  cvt_f32_to_f16_kernel<<<NW / (256 * 8), 256, 0, stream>>>(Wk, Wkh, NW);
  cvt_f32_to_f16_kernel<<<NW / (256 * 8), 256, 0, stream>>>(Wv, Wvh, NW);
  cvt_f32_to_f16_kernel<<<NW / (256 * 8), 256, 0, stream>>>(Wo, Woh, NW);

  dim3 ggrid(M / TM, N / TN);        // 32 x 6
  gemm_nt_bias_wmma<<<ggrid, 256, 0, stream>>>(xh, Wqh, bq, qb, M, N, K, qscale);
  gemm_nt_bias_wmma<<<ggrid, 256, 0, stream>>>(xh, Wkh, bk, kb, M, N, K, 1.0f);
  gemm_nt_bias_wmma<<<ggrid, 256, 0, stream>>>(xh, Wvh, bv, vb, M, N, K, 1.0f);

  local_select_kernel<<<dim3(BHTOT, T_SEQ / 64), 64, 0, stream>>>(qb, kb, sel);
  qp_kernel<<<1, N_HEADS * NPROTO, 0, stream>>>(qb, Qp);
  proto_score_kernel<<<dim3(N_HEADS, T_SEQ / 256), 256, 0, stream>>>(kb, Qp, S, u);
  greedy_kernel<<<N_HEADS, 256, 0, stream>>>(u, S, glob);
  attend_kernel<<<dim3(BHTOT, T_SEQ / 64), 64, 0, stream>>>(qb, kb, vb, sel, glob, ah);

  gemm_nt_bias_wmma<<<ggrid, 256, 0, stream>>>(ah, Woh, bo, out, M, N, K, 1.0f);
}
